// BlockSparseMoE_10222022165263
// MI455X (gfx1250) — matrix operand order repaired and checked by reading, not compiled, not run on hardware
//
#include <hip/hip_runtime.h>
#include <hip/hip_bf16.h>

typedef __attribute__((ext_vector_type(16))) __bf16        v16bf;
typedef __attribute__((ext_vector_type(8)))  float         v8f;
typedef __attribute__((ext_vector_type(4)))  unsigned int  u32x4;

#define HDIM 1024
#define FDIM 2048
#define NEXP 8
#define MT   32      // tokens per workgroup tile
#define PX   1032    // LDS pitch for x tile (bf16 elems, 16B aligned rows, bank-skewed)
#define PH   136     // LDS pitch for h tile

static __device__ __forceinline__ unsigned short f2bf(float f) {
  unsigned int u = __builtin_bit_cast(unsigned int, f);
  u += 0x7FFFu + ((u >> 16) & 1u);          // round-to-nearest-even
  return (unsigned short)(u >> 16);
}
static __device__ __forceinline__ unsigned int pack2(float a, float b) {
  return (unsigned int)f2bf(a) | ((unsigned int)f2bf(b) << 16);
}

union BFrag {
  u32x4 q[2];
  v16bf v;
};

static __device__ __forceinline__ v8f wmma_bf16(v16bf a, v16bf b, v8f c) {
  // D = A(16x32) * B(32x16) + C, fp32 accumulate
  return __builtin_amdgcn_wmma_f32_16x16x32_bf16(false, a, false, b, (short)0, c,
                                                 false, false);
}

// B fragment: lane holds column n, K = 16*(lane>=16) + 0..15 contiguous.
template <bool WBF16>
static __device__ __forceinline__ BFrag load_bfrag(const void* __restrict__ base,
                                                   size_t rowOff, int kb) {
  BFrag b;
  if constexpr (WBF16) {
    const unsigned short* p = (const unsigned short*)base + rowOff + kb;
    b.q[0] = *(const u32x4*)(p);
    b.q[1] = *(const u32x4*)(p + 8);
  } else {
    const float* p = (const float*)base + rowOff + kb;
    float4 f0 = *(const float4*)(p);
    float4 f1 = *(const float4*)(p + 4);
    float4 f2 = *(const float4*)(p + 8);
    float4 f3 = *(const float4*)(p + 12);
    u32x4 q0, q1;
    q0.x = pack2(f0.x, f0.y); q0.y = pack2(f0.z, f0.w);
    q0.z = pack2(f1.x, f1.y); q0.w = pack2(f1.z, f1.w);
    q1.x = pack2(f2.x, f2.y); q1.y = pack2(f2.z, f2.w);
    q1.z = pack2(f3.x, f3.y); q1.w = pack2(f3.z, f3.w);
    b.q[0] = q0; b.q[1] = q1;
  }
  return b;
}

// ---------------- fp32 -> bf16 conversion ----------------
__global__ __launch_bounds__(256) void cvt_kernel(const float* __restrict__ s,
                                                  unsigned short* __restrict__ d,
                                                  long n) {
  long i = ((long)blockIdx.x * 256 + threadIdx.x) * 4;
  if (i >= n) return;
  float4 v = *(const float4*)(s + i);
  unsigned int* dp = (unsigned int*)(d + i);
  dp[0] = pack2(v.x, v.y);
  dp[1] = pack2(v.z, v.w);
}

// ---------------- router: softmax top-2, build per-expert lists ----------------
__global__ __launch_bounds__(256) void router_kernel(const float* __restrict__ x,
                                                     const float* __restrict__ gw,
                                                     int T, int* __restrict__ counts,
                                                     int* __restrict__ lists,
                                                     float* __restrict__ wl) {
  __shared__ float gs[NEXP * HDIM];
  for (int i = threadIdx.x; i < NEXP * HDIM; i += 256) gs[i] = gw[i];
  __syncthreads();

  int lane = threadIdx.x & 31;
  int wave = threadIdx.x >> 5;
  int t = blockIdx.x * 8 + wave;
  if (t >= T) return;

  float acc[NEXP];
#pragma unroll
  for (int e = 0; e < NEXP; ++e) acc[e] = 0.f;
  const float* xr = x + (size_t)t * HDIM;
#pragma unroll
  for (int it = 0; it < HDIM / 128; ++it) {
    int h = it * 128 + lane * 4;
    float4 xv = *(const float4*)(xr + h);
#pragma unroll
    for (int e = 0; e < NEXP; ++e) {
      const float* g = gs + e * HDIM + h;
      acc[e] += xv.x * g[0] + xv.y * g[1] + xv.z * g[2] + xv.w * g[3];
    }
  }
#pragma unroll
  for (int off = 16; off > 0; off >>= 1)
#pragma unroll
    for (int e = 0; e < NEXP; ++e) acc[e] += __shfl_down(acc[e], off, 32);

  if (lane == 0) {
    int e1 = 0;
#pragma unroll
    for (int e = 1; e < NEXP; ++e) if (acc[e] > acc[e1]) e1 = e;
    int e2 = (e1 == 0) ? 1 : 0;
#pragma unroll
    for (int e = 0; e < NEXP; ++e)
      if (e != e1 && acc[e] > acc[e2]) e2 = e;
    float p2 = __expf(acc[e2] - acc[e1]);
    float inv = 1.f / (1.f + p2);
    int p = atomicAdd(&counts[e1], 1);
    lists[(size_t)e1 * T + p] = t;
    wl[(size_t)e1 * T + p] = inv;
    p = atomicAdd(&counts[e2], 1);
    lists[(size_t)e2 * T + p] = t;
    wl[(size_t)e2 * T + p] = p2 * inv;
  }
}

// ---------------- fused SwiGLU expert GEMM (bf16 WMMA) ----------------
// WBF16 = weights pre-converted to bf16 in ws (and x pre-converted, staged via
// async-to-LDS). Otherwise fp32 weights converted in registers.
template <bool WBF16>
__global__ __launch_bounds__(256) void moe_kernel(
    const float* __restrict__ x,
    const unsigned short* __restrict__ xbf,
    const void* __restrict__ w1v,
    const void* __restrict__ w3v,
    const void* __restrict__ w2v,
    const int* __restrict__ lists, const float* __restrict__ wl,
    const int* __restrict__ counts, float* __restrict__ out,
    int T, int tilesPer) {
  extern __shared__ char smem[];
  unsigned short* xs = (unsigned short*)smem;                        // MT*PX bf16
  unsigned short* hs = (unsigned short*)(smem + MT * PX * 2);        // MT*PH bf16
  int*   toks = (int*)(smem + MT * PX * 2 + MT * PH * 2);
  float* tws  = (float*)(toks + MT);

  int e = blockIdx.x / tilesPer;
  int tile = blockIdx.x % tilesPer;
  int cnt = counts[e];
  if (tile * MT >= cnt) return;

  int tid = threadIdx.x;
  if (tid < MT) {
    int idx = tile * MT + tid;
    int cl = idx < cnt ? idx : cnt - 1;            // pad with dup token, weight 0
    toks[tid] = lists[(size_t)e * T + cl];
    tws[tid] = idx < cnt ? wl[(size_t)e * T + idx] : 0.f;
  }
  __syncthreads();

  { // gather MT x rows -> LDS bf16, 8 threads per row
    int r = tid >> 3;
    int c0 = (tid & 7) * 128;
    if constexpr (WBF16) {
      // async DMA of pre-converted bf16 rows straight into LDS (ASYNCcnt path)
      const unsigned short* src = xbf + (size_t)toks[r] * HDIM + c0;
      unsigned int lbase =
          (unsigned int)(size_t)(xs + r * PX + c0);  // LDS byte offset
#pragma unroll
      for (int i = 0; i < 16; ++i) {                 // 16 x 16B chunks = 128 elems
        unsigned int laddr = lbase + i * 16u;
        unsigned long long ga = (unsigned long long)(size_t)(src + i * 8);
        asm volatile("global_load_async_to_lds_b128 %0, %1, off"
                     :: "v"(laddr), "v"(ga) : "memory");
      }
      asm volatile("s_wait_asynccnt 0" ::: "memory");
    } else {
      const float* xr = x + (size_t)toks[r] * HDIM + c0;
      unsigned int* dst = (unsigned int*)(xs + r * PX + c0);
#pragma unroll 8
      for (int i = 0; i < 32; ++i) {
        float4 v = *(const float4*)(xr + i * 4);
        dst[i * 2 + 0] = pack2(v.x, v.y);
        dst[i * 2 + 1] = pack2(v.z, v.w);
      }
    }
  }
  __syncthreads();

  int lane = tid & 31, wave = tid >> 5;
  int l15 = lane & 15, lh = lane >> 4;

  v8f zero = {};
  v8f acc[2][8];
#pragma unroll
  for (int a = 0; a < 2; ++a)
#pragma unroll
    for (int b = 0; b < 8; ++b) acc[a][b] = zero;

  for (int j = 0; j < FDIM / 128; ++j) {
    // -------- phase 1: h = silu(x@w1^T) * (x@w3^T), MTx128 chunk, 16 cols/wave
    v8f c1[2] = {zero, zero}, c3[2] = {zero, zero};
    int nrow = j * 128 + wave * 16 + l15;          // ffn column handled by this lane
    size_t row13 = ((size_t)e * FDIM + nrow) * HDIM;
#pragma unroll 4
    for (int k = 0; k < HDIM; k += 32) {
      int ka = k + (lh << 4);                      // A frag K base (16-contig)
      int kb = k + (lh << 4);                      // B frag K base (16-contig)
      BFrag a0, a1;
      a0.q[0] = *(const u32x4*)(xs + l15 * PX + ka);
      a0.q[1] = *(const u32x4*)(xs + l15 * PX + ka + 8);
      a1.q[0] = *(const u32x4*)(xs + (l15 + 16) * PX + ka);
      a1.q[1] = *(const u32x4*)(xs + (l15 + 16) * PX + ka + 8);
      BFrag b1 = load_bfrag<WBF16>(w1v, row13, kb);
      BFrag b3 = load_bfrag<WBF16>(w3v, row13, kb);
      c1[0] = wmma_bf16(a0.v, b1.v, c1[0]);
      c1[1] = wmma_bf16(a1.v, b1.v, c1[1]);
      c3[0] = wmma_bf16(a0.v, b3.v, c3[0]);
      c3[1] = wmma_bf16(a1.v, b3.v, c3[1]);
    }
#pragma unroll
    for (int mt = 0; mt < 2; ++mt)
#pragma unroll
      for (int i = 0; i < 8; ++i) {
        float g = c1[mt][i];
        float hv = (g / (1.f + __expf(-g))) * c3[mt][i];   // silu(g) * h3
        int m = mt * 16 + (lh << 3) + i;                    // C layout row
        hs[m * PH + wave * 16 + l15] = f2bf(hv);
      }
    __syncthreads();

    // -------- phase 2: out += h @ w2^T, each wave owns 128 output cols
#pragma unroll
    for (int k = 0; k < 128; k += 32) {
      int ka = k + (lh << 4);
      int kb = k + (lh << 4);
      BFrag a0, a1;
      a0.q[0] = *(const u32x4*)(hs + l15 * PH + ka);
      a0.q[1] = *(const u32x4*)(hs + l15 * PH + ka + 8);
      a1.q[0] = *(const u32x4*)(hs + (l15 + 16) * PH + ka);
      a1.q[1] = *(const u32x4*)(hs + (l15 + 16) * PH + ka + 8);
#pragma unroll
      for (int nt = 0; nt < 8; ++nt) {
        int o = wave * 128 + nt * 16 + l15;
        size_t row2 = ((size_t)e * HDIM + o) * FDIM + j * 128;
        BFrag b = load_bfrag<WBF16>(w2v, row2, kb);
        acc[0][nt] = wmma_bf16(a0.v, b.v, acc[0][nt]);
        acc[1][nt] = wmma_bf16(a1.v, b.v, acc[1][nt]);
      }
    }
    __syncthreads();   // hs reused next j
  }

  // -------- epilogue: scale by routing weight, atomic accumulate
#pragma unroll
  for (int mt = 0; mt < 2; ++mt)
#pragma unroll
    for (int nt = 0; nt < 8; ++nt)
#pragma unroll
      for (int i = 0; i < 8; ++i) {
        int m = mt * 16 + (lh << 3) + i;
        int o = wave * 128 + nt * 16 + l15;
        atomicAdd(out + (size_t)toks[m] * HDIM + o, acc[mt][nt][i] * tws[m]);
      }
}

extern "C" void kernel_launch(void* const* d_in, const int* in_sizes, int n_in,
                              void* d_out, int out_size, void* d_ws, size_t ws_size,
                              hipStream_t stream) {
  const float* x  = (const float*)d_in[0];
  const float* gw = (const float*)d_in[1];
  const float* w1 = (const float*)d_in[2];
  const float* w2 = (const float*)d_in[3];   // dict order: w1, w2, w3
  const float* w3 = (const float*)d_in[4];
  long T = (long)in_sizes[0] / HDIM;

  size_t welems = (size_t)NEXP * FDIM * HDIM;     // 16M elems per weight matrix
  size_t xelems = (size_t)T * HDIM;
  size_t routeBytes = (size_t)NEXP * T * 8 + 256;
  size_t needBig = (3 * welems + xelems) * 2 + routeBytes;
  bool big = ws_size >= needBig;

  unsigned short *w1b = nullptr, *w3b = nullptr, *w2b = nullptr, *xbf = nullptr;
  char* cur = (char*)d_ws;
  if (big) {
    w1b = (unsigned short*)cur; cur += welems * 2;
    w3b = (unsigned short*)cur; cur += welems * 2;
    w2b = (unsigned short*)cur; cur += welems * 2;
    xbf = (unsigned short*)cur; cur += xelems * 2;
  }
  int*   lists  = (int*)cur;
  float* wl     = (float*)(lists + (size_t)NEXP * T);
  int*   counts = (int*)(wl + (size_t)NEXP * T);

  hipMemsetAsync(d_out, 0, (size_t)out_size * sizeof(float), stream);
  hipMemsetAsync(counts, 0, NEXP * sizeof(int), stream);

  if (big) {
    long n = (long)welems;
    int cb = (int)((n / 4 + 255) / 256);
    cvt_kernel<<<cb, 256, 0, stream>>>(w1, w1b, n);
    cvt_kernel<<<cb, 256, 0, stream>>>(w3, w3b, n);
    cvt_kernel<<<cb, 256, 0, stream>>>(w2, w2b, n);
    long nx = (long)xelems;
    cvt_kernel<<<(int)((nx / 4 + 255) / 256), 256, 0, stream>>>(x, xbf, nx);
  }

  router_kernel<<<(int)((T + 7) / 8), 256, 0, stream>>>(x, gw, (int)T, counts, lists, wl);

  int tilesPer = (int)((T + MT - 1) / MT);
  size_t shmem = (size_t)MT * PX * 2 + (size_t)MT * PH * 2 + MT * 4 + MT * 4; // 75008 B
  if (big) {
    hipFuncSetAttribute((const void*)moe_kernel<true>,
                        hipFuncAttributeMaxDynamicSharedMemorySize, (int)shmem);
    moe_kernel<true><<<NEXP * tilesPer, 256, shmem, stream>>>(
        x, xbf, w1b, w3b, w2b, lists, wl, counts, (float*)d_out, (int)T, tilesPer);
  } else {
    hipFuncSetAttribute((const void*)moe_kernel<false>,
                        hipFuncAttributeMaxDynamicSharedMemorySize, (int)shmem);
    moe_kernel<false><<<NEXP * tilesPer, 256, shmem, stream>>>(
        x, nullptr, w1, w3, w2, lists, wl, counts, (float*)d_out, (int)T, tilesPer);
  }
}
